// Net_66090956751514
// MI455X (gfx1250) — matrix operand order repaired and checked by reading, not compiled, not run on hardware
//
#include <hip/hip_runtime.h>
#include <hip/hip_bf16.h>
#include <math.h>
#include <float.h>

typedef _Float16 f16;
typedef __attribute__((ext_vector_type(16))) _Float16 v16h;
typedef __attribute__((ext_vector_type(8)))  float    v8f;

#define F_IN   256
#define NHEAD  8
#define CH1    8
#define F_MID  64      // 8*8
#define NCLS   40
#define NCLS_P 48      // padded to 3 WMMA tiles
#define LDS_LD 40      // padded LDS row stride (f16 units): 80B = 5x16B, bank-conflict free
#define NEG_SLOPE 0.2f

// ---------------------------------------------------------------- helpers

static __device__ __forceinline__ float leaky(float v) {
    return v > 0.0f ? v : NEG_SLOPE * v;
}

static __device__ __forceinline__ void atomicMaxF32(float* addr, float val) {
    unsigned int* ai = (unsigned int*)addr;
    unsigned int old = *ai;
    while (__uint_as_float(old) < val) {
        unsigned int assumed = old;
        old = atomicCAS(ai, assumed, __float_as_uint(val));
        if (old == assumed) break;
    }
}

// WMMA fragment loaders (CDNA5 ISA 7.12.2, wave32)
// A 16x32 f16, A-tile row-major in LDS (row stride LDS_LD):
// lane m=l&15, half=l>>4; a[e]=K(e+16*half), e=0..15
static __device__ __forceinline__ v16h load_frag_a(const f16* As, int rowBase, int lane) {
    int m = lane & 15, half = lane >> 4;
    const f16* rowp = As + (rowBase + m) * LDS_LD;
    v16h a;
#pragma unroll
    for (int v = 0; v < 8; ++v) {
        int k0 = 2 * v + 16 * half;
        a[2 * v]     = rowp[k0];
        a[2 * v + 1] = rowp[k0 + 1];
    }
    return a;
}

// B 32x16 f16, B-tile stored N-major in LDS (BsT[n][k], row stride LDS_LD):
// lane n=l&15, half=l>>4; element e: K = e + 16*half -> 16 contiguous f16 (2x b128)
static __device__ __forceinline__ v16h load_frag_bT(const f16* BsT, int colBase, int lane) {
    int n = lane & 15, half = lane >> 4;
    const f16* p = BsT + (colBase + n) * LDS_LD + 16 * half;
    v16h b;
#pragma unroll
    for (int e = 0; e < 16; ++e) b[e] = p[e];
    return b;
}

// ---------------------------------------------------------------- utility kernels

__global__ void cvt_f32_f16_kernel(const float* __restrict__ src, f16* __restrict__ dst, int n) {
    int i = blockIdx.x * blockDim.x + threadIdx.x;
    if (i < n) dst[i] = (f16)src[i];
}

// W1 [256,64] f32 -> W1^T [64,256] f16 (N-major)
__global__ void cvt_w1t_kernel(const float* __restrict__ src, f16* __restrict__ dst) {
    int i = blockIdx.x * blockDim.x + threadIdx.x;   // over 64*256
    if (i >= F_MID * F_IN) return;
    int n = i / F_IN, k = i % F_IN;
    dst[i] = (f16)src[(size_t)k * F_MID + n];
}

// W2 [64,40] f32 -> W2^T [48,64] f16 zero-padded (N-major)
__global__ void cvt_w2t_kernel(const float* __restrict__ src, f16* __restrict__ dst) {
    int i = blockIdx.x * blockDim.x + threadIdx.x;   // over 48*64
    if (i >= NCLS_P * F_MID) return;
    int n = i / F_MID, k = i % F_MID;
    dst[i] = (n < NCLS) ? (f16)src[(size_t)k * NCLS + n] : (f16)0.0f;
}

__global__ void fill_f32_kernel(float* __restrict__ p, float v, int n) {
    int i = blockIdx.x * blockDim.x + threadIdx.x;
    if (i < n) p[i] = v;
}

// ---------------------------------------------------------------- GEMM1: [N,256] x W1 -> f32 [N,64]
// block: 256 threads = 8 waves; block tile 32x64; wave tile 16x16; K step 32
// A row-major f16, B pre-transposed (N-major) f16
__global__ void __launch_bounds__(256) gemm1_kernel(const f16* __restrict__ A,
                                                    const f16* __restrict__ BT,
                                                    float* __restrict__ C) {
    __shared__ f16 As[32 * LDS_LD];
    __shared__ f16 BsT[64 * LDS_LD];
    const int tid = threadIdx.x;
    const int wave = tid >> 5, lane = tid & 31;
    const int rowBlock = blockIdx.x * 32;
    const int tileM = (wave >> 2) * 16;   // 0 / 16
    const int tileN = (wave & 3) * 16;    // 0..48
    v8f acc = {};
    for (int kt = 0; kt < F_IN; kt += 32) {
        {   // A tile 32 rows x 32 k: 4 f16 (8B) per thread
            int r = tid >> 3, c = (tid & 7) * 4;
            *(uint2*)(As + r * LDS_LD + c) =
                *(const uint2*)(A + (size_t)(rowBlock + r) * F_IN + kt + c);
        }
        {   // B tile 64 n x 32 k (N-major): 8 f16 (16B) per thread
            int n = tid >> 2, kc = (tid & 3) * 8;
            *(uint4*)(BsT + n * LDS_LD + kc) =
                *(const uint4*)(BT + (size_t)n * F_IN + kt + kc);
        }
        __syncthreads();
        v16h a = load_frag_a(As, tileM, lane);
        v16h b = load_frag_bT(BsT, tileN, lane);
        acc = __builtin_amdgcn_wmma_f32_16x16x32_f16(false, a, false, b, (short)0, acc, false, false);
        __syncthreads();
    }
    int n = lane & 15, half = lane >> 4;
#pragma unroll
    for (int r = 0; r < 8; ++r) {
        int row = rowBlock + tileM + r + 8 * half;
        C[(size_t)row * F_MID + tileN + n] = acc[r];
    }
}

// ---------------------------------------------------------------- GEMM2: [N,64] x W2 -> f32 [N,40]
__global__ void __launch_bounds__(256) gemm2_kernel(const f16* __restrict__ A,
                                                    const f16* __restrict__ BT,
                                                    float* __restrict__ C) {
    __shared__ f16 As[32 * LDS_LD];
    __shared__ f16 BsT[NCLS_P * LDS_LD];
    const int tid = threadIdx.x;
    const int wave = tid >> 5, lane = tid & 31;
    const int rowBlock = blockIdx.x * 32;
    const int tileM = (wave / 3) * 16;    // waves 0..5 compute
    const int tileN = (wave % 3) * 16;
    v8f acc = {};
    for (int kt = 0; kt < F_MID; kt += 32) {
        {   // A tile 32 x 32
            int r = tid >> 3, c = (tid & 7) * 4;
            *(uint2*)(As + r * LDS_LD + c) =
                *(const uint2*)(A + (size_t)(rowBlock + r) * F_MID + kt + c);
        }
        if (tid < 192) {   // B tile 48 n x 32 k (N-major): 8 f16 per thread
            int n = tid >> 2, kc = (tid & 3) * 8;
            *(uint4*)(BsT + n * LDS_LD + kc) =
                *(const uint4*)(BT + (size_t)n * F_MID + kt + kc);
        }
        __syncthreads();
        if (wave < 6) {
            v16h a = load_frag_a(As, tileM, lane);
            v16h b = load_frag_bT(BsT, tileN, lane);
            acc = __builtin_amdgcn_wmma_f32_16x16x32_f16(false, a, false, b, (short)0, acc, false, false);
        }
        __syncthreads();
    }
    if (wave < 6) {
        int n = lane & 15, half = lane >> 4;
#pragma unroll
        for (int r = 0; r < 8; ++r) {
            int col = tileN + n;
            int row = rowBlock + tileM + r + 8 * half;
            if (col < NCLS) C[(size_t)row * NCLS + col] = acc[r];
        }
    }
}

// ---------------------------------------------------------------- attention logits
__global__ void alpha1_kernel(const float* __restrict__ h1,
                              const float* __restrict__ att_s, const float* __restrict__ att_d,
                              float* __restrict__ as, float* __restrict__ ad, int NH) {
    int i = blockIdx.x * blockDim.x + threadIdx.x;
    if (i >= NH) return;
    int hd = i & (NHEAD - 1);
    const float* hp = h1 + (size_t)i * CH1;   // n*64 + h*8 == i*8
    float s = 0.f, d = 0.f;
#pragma unroll
    for (int c = 0; c < CH1; ++c) {
        float v = hp[c];
        s += v * att_s[hd * CH1 + c];
        d += v * att_d[hd * CH1 + c];
    }
    as[i] = s; ad[i] = d;
}

__global__ void alpha2_kernel(const float* __restrict__ h2,
                              const float* __restrict__ att_s, const float* __restrict__ att_d,
                              float* __restrict__ as, float* __restrict__ ad, int N) {
    int n = blockIdx.x * blockDim.x + threadIdx.x;
    if (n >= N) return;
    const float* hp = h2 + (size_t)n * NCLS;
    float s = 0.f, d = 0.f;
#pragma unroll
    for (int c = 0; c < NCLS; ++c) {
        float v = hp[c];
        s += v * att_s[c];
        d += v * att_d[c];
    }
    as[n] = s; ad[n] = d;
}

// ---------------------------------------------------------------- edge passes, layer 1 (8 heads)
__global__ void edge_max1_kernel(const int* __restrict__ src, const int* __restrict__ dst,
                                 const float* __restrict__ as, const float* __restrict__ ad,
                                 float* __restrict__ m, int E) {
    int e = blockIdx.x * blockDim.x + threadIdx.x;
    if (e >= E) return;
    int s = src[e], d = dst[e];
#pragma unroll
    for (int h = 0; h < NHEAD; ++h) {
        float v = leaky(as[(size_t)s * NHEAD + h] + ad[(size_t)d * NHEAD + h]);
        atomicMaxF32(&m[(size_t)d * NHEAD + h], v);
    }
}

__global__ void edge_den1_kernel(const int* __restrict__ src, const int* __restrict__ dst,
                                 const float* __restrict__ as, const float* __restrict__ ad,
                                 const float* __restrict__ m, float* __restrict__ den, int E) {
    int e = blockIdx.x * blockDim.x + threadIdx.x;
    if (e >= E) return;
    int s = src[e], d = dst[e];
#pragma unroll
    for (int h = 0; h < NHEAD; ++h) {
        float v = leaky(as[(size_t)s * NHEAD + h] + ad[(size_t)d * NHEAD + h]);
        atomicAdd(&den[(size_t)d * NHEAD + h], __expf(v - m[(size_t)d * NHEAD + h]));
    }
}

__global__ void edge_msg1_kernel(const int* __restrict__ src, const int* __restrict__ dst,
                                 const float* __restrict__ as, const float* __restrict__ ad,
                                 const float* __restrict__ m, const float* __restrict__ den,
                                 const float* __restrict__ h1, float* __restrict__ agg, int E) {
    int e = blockIdx.x * blockDim.x + threadIdx.x;
    if (e >= E) return;
    int s = src[e], d = dst[e];
    float w[NHEAD];
#pragma unroll
    for (int h = 0; h < NHEAD; ++h) {
        float v = leaky(as[(size_t)s * NHEAD + h] + ad[(size_t)d * NHEAD + h]);
        w[h] = __expf(v - m[(size_t)d * NHEAD + h]) / den[(size_t)d * NHEAD + h];
    }
    const float* hs = h1 + (size_t)s * F_MID;
    float* ao = agg + (size_t)d * F_MID;
#pragma unroll
    for (int j = 0; j < F_MID; ++j) {
        atomicAdd(&ao[j], w[j >> 3] * hs[j]);
    }
}

// ---------------------------------------------------------------- edge passes, layer 2 (1 head, 40 ch)
__global__ void edge_max2_kernel(const int* __restrict__ src, const int* __restrict__ dst,
                                 const float* __restrict__ as, const float* __restrict__ ad,
                                 float* __restrict__ m, int E) {
    int e = blockIdx.x * blockDim.x + threadIdx.x;
    if (e >= E) return;
    atomicMaxF32(&m[dst[e]], leaky(as[src[e]] + ad[dst[e]]));
}

__global__ void edge_den2_kernel(const int* __restrict__ src, const int* __restrict__ dst,
                                 const float* __restrict__ as, const float* __restrict__ ad,
                                 const float* __restrict__ m, float* __restrict__ den, int E) {
    int e = blockIdx.x * blockDim.x + threadIdx.x;
    if (e >= E) return;
    int d = dst[e];
    float v = leaky(as[src[e]] + ad[d]);
    atomicAdd(&den[d], __expf(v - m[d]));
}

__global__ void edge_msg2_kernel(const int* __restrict__ src, const int* __restrict__ dst,
                                 const float* __restrict__ as, const float* __restrict__ ad,
                                 const float* __restrict__ m, const float* __restrict__ den,
                                 const float* __restrict__ h2, float* __restrict__ agg, int E) {
    int e = blockIdx.x * blockDim.x + threadIdx.x;
    if (e >= E) return;
    int s = src[e], d = dst[e];
    float v = leaky(as[s] + ad[d]);
    float w = __expf(v - m[d]) / den[d];
    const float* hs = h2 + (size_t)s * NCLS;
    float* ao = agg + (size_t)d * NCLS;
#pragma unroll
    for (int c = 0; c < NCLS; ++c) atomicAdd(&ao[c], w * hs[c]);
}

// ---------------------------------------------------------------- fused bias + ELU + f16 convert
__global__ void bias_elu_f16_kernel(const float* __restrict__ agg, const float* __restrict__ b,
                                    f16* __restrict__ out, int total) {
    int i = blockIdx.x * blockDim.x + threadIdx.x;
    if (i >= total) return;
    float v = agg[i] + b[i & (F_MID - 1)];
    v = v > 0.f ? v : (__expf(v) - 1.0f);
    out[i] = (f16)v;
}

// ---------------------------------------------------------------- bias + log_softmax (40 classes)
__global__ void logsoftmax_kernel(const float* __restrict__ agg, const float* __restrict__ b,
                                  float* __restrict__ out, int N) {
    int n = blockIdx.x * blockDim.x + threadIdx.x;
    if (n >= N) return;
    const float* ap = agg + (size_t)n * NCLS;
    float vals[NCLS];
    float mx = -FLT_MAX;
#pragma unroll
    for (int c = 0; c < NCLS; ++c) {
        float v = ap[c] + b[c];
        vals[c] = v;
        mx = fmaxf(mx, v);
    }
    float sum = 0.f;
#pragma unroll
    for (int c = 0; c < NCLS; ++c) sum += __expf(vals[c] - mx);
    float lse = mx + __logf(sum);
    float* op = out + (size_t)n * NCLS;
#pragma unroll
    for (int c = 0; c < NCLS; ++c) op[c] = vals[c] - lse;
}

// ---------------------------------------------------------------- launch

extern "C" void kernel_launch(void* const* d_in, const int* in_sizes, int n_in,
                              void* d_out, int out_size, void* d_ws, size_t ws_size,
                              hipStream_t stream) {
    const float* x       = (const float*)d_in[0];
    const float* W1      = (const float*)d_in[1];
    const float* att_s1  = (const float*)d_in[2];
    const float* att_d1  = (const float*)d_in[3];
    const float* b1      = (const float*)d_in[4];
    const float* W2      = (const float*)d_in[5];
    const float* att_s2  = (const float*)d_in[6];
    const float* att_d2  = (const float*)d_in[7];
    const float* b2      = (const float*)d_in[8];
    const int*   ei      = (const int*)d_in[9];

    const int N = in_sizes[0] / F_IN;          // 100000
    const int E = in_sizes[9] / 2;             // 1,700,000 (with self loops)
    const int* srcI = ei;
    const int* dstI = ei + E;

    // workspace carve (256B aligned)
    char* p = (char*)d_ws;
    auto alloc = [&](size_t bytes) -> void* {
        void* r = (void*)p;
        p += (bytes + 255) & ~(size_t)255;
        return r;
    };
    f16*   xbf   = (f16*)  alloc((size_t)N * F_IN * sizeof(f16));
    f16*   w1t   = (f16*)  alloc((size_t)F_MID * F_IN * sizeof(f16));    // W1^T [64,256]
    f16*   w2t   = (f16*)  alloc((size_t)NCLS_P * F_MID * sizeof(f16));  // W2^T [48,64]
    float* h1    = (float*)alloc((size_t)N * F_MID * sizeof(float));
    float* as1   = (float*)alloc((size_t)N * NHEAD * sizeof(float));
    float* ad1   = (float*)alloc((size_t)N * NHEAD * sizeof(float));
    float* m1    = (float*)alloc((size_t)N * NHEAD * sizeof(float));
    float* dn1   = (float*)alloc((size_t)N * NHEAD * sizeof(float));
    float* agg1  = (float*)alloc((size_t)N * F_MID * sizeof(float));
    f16*   h1bf  = (f16*)  alloc((size_t)N * F_MID * sizeof(f16));
    float* h2    = (float*)alloc((size_t)N * NCLS * sizeof(float));
    float* as2   = (float*)alloc((size_t)N * sizeof(float));
    float* ad2   = (float*)alloc((size_t)N * sizeof(float));
    float* m2    = (float*)alloc((size_t)N * sizeof(float));
    float* dn2   = (float*)alloc((size_t)N * sizeof(float));
    float* agg2  = (float*)alloc((size_t)N * NCLS * sizeof(float));
    (void)ws_size; (void)n_in; (void)out_size;

    const int TB = 256;
    auto blocks = [&](long long n) { return dim3((unsigned)((n + TB - 1) / TB)); };

    // --- precision converts / weight transposes
    cvt_f32_f16_kernel<<<blocks((long long)N * F_IN), TB, 0, stream>>>(x, xbf, N * F_IN);
    cvt_w1t_kernel<<<blocks(F_MID * F_IN), TB, 0, stream>>>(W1, w1t);
    cvt_w2t_kernel<<<blocks(NCLS_P * F_MID), TB, 0, stream>>>(W2, w2t);

    // --- layer 1
    gemm1_kernel<<<dim3(N / 32), TB, 0, stream>>>(xbf, w1t, h1);
    alpha1_kernel<<<blocks((long long)N * NHEAD), TB, 0, stream>>>(h1, att_s1, att_d1, as1, ad1, N * NHEAD);
    fill_f32_kernel<<<blocks((long long)N * NHEAD), TB, 0, stream>>>(m1, -FLT_MAX, N * NHEAD);
    fill_f32_kernel<<<blocks((long long)N * NHEAD), TB, 0, stream>>>(dn1, 0.f, N * NHEAD);
    fill_f32_kernel<<<blocks((long long)N * F_MID), TB, 0, stream>>>(agg1, 0.f, N * F_MID);
    edge_max1_kernel<<<blocks(E), TB, 0, stream>>>(srcI, dstI, as1, ad1, m1, E);
    edge_den1_kernel<<<blocks(E), TB, 0, stream>>>(srcI, dstI, as1, ad1, m1, dn1, E);
    edge_msg1_kernel<<<blocks(E), TB, 0, stream>>>(srcI, dstI, as1, ad1, m1, dn1, h1, agg1, E);
    bias_elu_f16_kernel<<<blocks((long long)N * F_MID), TB, 0, stream>>>(agg1, b1, h1bf, N * F_MID);

    // --- layer 2
    gemm2_kernel<<<dim3(N / 32), TB, 0, stream>>>(h1bf, w2t, h2);
    alpha2_kernel<<<blocks(N), TB, 0, stream>>>(h2, att_s2, att_d2, as2, ad2, N);
    fill_f32_kernel<<<blocks(N), TB, 0, stream>>>(m2, -FLT_MAX, N);
    fill_f32_kernel<<<blocks(N), TB, 0, stream>>>(dn2, 0.f, N);
    fill_f32_kernel<<<blocks((long long)N * NCLS), TB, 0, stream>>>(agg2, 0.f, N * NCLS);
    edge_max2_kernel<<<blocks(E), TB, 0, stream>>>(srcI, dstI, as2, ad2, m2, E);
    edge_den2_kernel<<<blocks(E), TB, 0, stream>>>(srcI, dstI, as2, ad2, m2, dn2, E);
    edge_msg2_kernel<<<blocks(E), TB, 0, stream>>>(srcI, dstI, as2, ad2, m2, dn2, h2, agg2, E);

    // --- output
    logsoftmax_kernel<<<blocks(N), TB, 0, stream>>>(agg2, b2, (float*)d_out, N);
}
